// AttentionBlock_86715389706345
// MI455X (gfx1250) — matrix operand order repaired and checked by reading, not compiled, not run on hardware
//
#include <hip/hip_runtime.h>
#include <hip/hip_bf16.h>
#include <stdint.h>

// Problem constants (fixed by the reference).
#define BB 8
#define TT 1024
#define DD 1024
#define HH 16
#define DHH 64
#define MM (BB * TT)   // 8192 rows for the FFN GEMMs
#define LN_EPS 1e-5f

typedef __attribute__((ext_vector_type(16))) _Float16 v16h;
typedef __attribute__((ext_vector_type(8)))  float    v8f;
typedef __attribute__((ext_vector_type(4)))  unsigned int u32x4;

// ---------------------------------------------------------------------------
// 1) xsum[b,c] = sum_t x[b,t,c]
__global__ void k_xsum(const float* __restrict__ x, float* __restrict__ xsum) {
    int idx = blockIdx.x * blockDim.x + threadIdx.x;     // 0 .. B*D-1
    int b = idx >> 10, c = idx & (DD - 1);
    const float* p = x + (size_t)b * TT * DD + c;
    float acc = 0.f;
    for (int t = 0; t < TT; ++t) acc += p[(size_t)t * DD];
    xsum[idx] = acc;
}

// 2) ksum[b,h,d] = xsum[b,:] @ Wk[h,:,d];  vsum likewise
__global__ void k_kvsum(const float* __restrict__ xsum,
                        const float* __restrict__ Wk, const float* __restrict__ Wv,
                        float* __restrict__ ksum, float* __restrict__ vsum) {
    int idx = blockIdx.x * blockDim.x + threadIdx.x;     // 0 .. B*H*DH-1
    int b = idx / (HH * DHH);
    int r = idx - b * (HH * DHH);
    int h = r / DHH, d = r - h * DHH;
    const float* wk = Wk + ((size_t)h * DD) * DHH + d;
    const float* wv = Wv + ((size_t)h * DD) * DHH + d;
    const float* xs = xsum + (size_t)b * DD;
    float ak = 0.f, av = 0.f;
    for (int c = 0; c < DD; ++c) {
        float xv = xs[c];
        ak += xv * wk[(size_t)c * DHH];
        av += xv * wv[(size_t)c * DHH];
    }
    ksum[idx] = ak;
    vsum[idx] = av;
}

// 3) wq_eff[b,h,c] = Wq[h,c,:] . ksum[b,h,:]
__global__ void k_wqeff(const float* __restrict__ Wq, const float* __restrict__ ksum,
                        float* __restrict__ wqeff) {
    int idx = blockIdx.x * blockDim.x + threadIdx.x;     // 0 .. B*H*D-1
    int b = idx >> 14;
    int r = idx & 16383;
    int h = r >> 10, c = r & 1023;
    const float* wq = Wq + ((size_t)h * DD + c) * DHH;
    const float* ks = ksum + ((size_t)b * HH + h) * DHH;
    float acc = 0.f;
#pragma unroll 8
    for (int d = 0; d < DHH; ++d) acc += wq[d] * ks[d];
    wqeff[idx] = acc;  // layout [(b*H+h)*D + c] == idx
}

// 4) rows[b,h,t] = (1/sqrt(DH)) * x[b,t,:] . wq_eff[b,h,:]   (one wave per dot)
__global__ void k_rowsum(const float* __restrict__ x, const float* __restrict__ wqeff,
                         float* __restrict__ rows) {
    int gw   = (blockIdx.x * blockDim.x + threadIdx.x) >> 5;  // global wave id
    int lane = threadIdx.x & 31;
    int t  = gw & (TT - 1);
    int bh = gw >> 10;                 // b*H + h
    int b  = bh >> 4;
    const float* xr = x + ((size_t)b * TT + t) * DD;
    const float* wr = wqeff + (size_t)bh * DD;
    float acc = 0.f;
    for (int i = lane; i < DD; i += 32) acc += xr[i] * wr[i];
#pragma unroll
    for (int off = 16; off > 0; off >>= 1) acc += __shfl_xor(acc, off, 32);
    if (lane == 0) rows[gw] = acc * 0.125f;   // 1/sqrt(64)
}

// 5) softmax over t, in place, one block per (b,h)
__global__ void k_softmax(float* __restrict__ rows) {
    __shared__ float red[256];
    float* r = rows + (size_t)blockIdx.x * TT;
    int tid = threadIdx.x;
    float lmax = -3.4e38f;
    for (int i = tid; i < TT; i += 256) lmax = fmaxf(lmax, r[i]);
    red[tid] = lmax; __syncthreads();
#pragma unroll
    for (int s = 128; s > 0; s >>= 1) { if (tid < s) red[tid] = fmaxf(red[tid], red[tid + s]); __syncthreads(); }
    float m = red[0]; __syncthreads();
    float lsum = 0.f;
    for (int i = tid; i < TT; i += 256) { float e = __expf(r[i] - m); r[i] = e; lsum += e; }
    red[tid] = lsum; __syncthreads();
#pragma unroll
    for (int s = 128; s > 0; s >>= 1) { if (tid < s) red[tid] += red[tid + s]; __syncthreads(); }
    float inv = 1.0f / red[0];
    for (int i = tid; i < TT; i += 256) r[i] *= inv;
}

__device__ __forceinline__ float block_sum256(float v, float* red) {
    int tid = threadIdx.x;
    red[tid] = v; __syncthreads();
#pragma unroll
    for (int s = 128; s > 0; s >>= 1) { if (tid < s) red[tid] += red[tid + s]; __syncthreads(); }
    float r = red[0]; __syncthreads();
    return r;
}

// 6) x1 = LN(x + out_attn), write f32 (residual) and f16 (WMMA A). One block per (b,t).
__global__ void k_attn_ln(const float* __restrict__ x, const float* __restrict__ p,
                          const float* __restrict__ vsum,
                          const float* __restrict__ g1, const float* __restrict__ b1,
                          float* __restrict__ x1f, _Float16* __restrict__ x1h) {
    __shared__ float red[256];
    int bt = blockIdx.x;
    int b = bt >> 10, t = bt & (TT - 1);
    const float* xr = x + (size_t)bt * DD;
    float v[4];
#pragma unroll
    for (int j = 0; j < 4; ++j) {
        int c = threadIdx.x + j * 256;
        int h = c >> 6, d = c & 63;
        int bh = b * HH + h;
        v[j] = xr[c] + p[(size_t)bh * TT + t] * vsum[(size_t)bh * DHH + d];
    }
    float mean = block_sum256(v[0] + v[1] + v[2] + v[3], red) * (1.f / DD);
    float q = 0.f;
#pragma unroll
    for (int j = 0; j < 4; ++j) { float dd = v[j] - mean; q += dd * dd; }
    float var = block_sum256(q, red) * (1.f / DD);
    float sc = rsqrtf(var + LN_EPS);
#pragma unroll
    for (int j = 0; j < 4; ++j) {
        int c = threadIdx.x + j * 256;
        float y = (v[j] - mean) * sc * g1[c] + b1[c];
        x1f[(size_t)bt * DD + c] = y;
        x1h[(size_t)bt * DD + c] = (_Float16)y;
    }
}

// 7) Wt[n,k] = (f16) W[k,n]  — tiled smem transpose + convert
__global__ void k_trans_h(const float* __restrict__ W, _Float16* __restrict__ Wt) {
    __shared__ float tile[32][33];
    int nbase = blockIdx.x * 32, kbase = blockIdx.y * 32;
    int tx = threadIdx.x, ty = threadIdx.y;  // (32,8)
#pragma unroll
    for (int i = 0; i < 4; ++i)
        tile[ty + i * 8][tx] = W[(size_t)(kbase + ty + i * 8) * DD + nbase + tx];
    __syncthreads();
#pragma unroll
    for (int i = 0; i < 4; ++i)
        Wt[(size_t)(nbase + ty + i * 8) * DD + kbase + tx] = (_Float16)tile[tx][ty + i * 8];
}

// 8) WMMA GEMM: C(MxN) = A(MxK,f16,row-major) @ Bt(NxK,f16,row-major)^T
//    Block tile 128x128, K-step 32, 8 waves: each wave owns a 32x64 C tile
//    -> 8 x v_wmma_f32_16x16x32_f16 per K-step. LDS double-buffered, 80B rows.
#define LDSS 40  // halves per LDS row (80 bytes: 16B aligned, conflict-free over 16 rows)
#define GK DD
#define GN DD

template <int RELU_F16_OUT>
__global__ __launch_bounds__(256) void k_gemm(const _Float16* __restrict__ A,
                                              const _Float16* __restrict__ Bt,
                                              void* __restrict__ Cout) {
    __shared__ alignas(16) _Float16 sA[2][128 * LDSS];
    __shared__ alignas(16) _Float16 sB[2][128 * LDSS];
    const int tid = threadIdx.x;
    const int m0 = blockIdx.y * 128, n0 = blockIdx.x * 128;

    const int lrow = tid >> 2;      // 0..63 (chunk 0), +64 for chunk 1
    const int lseg = (tid & 3) * 8; // half offset within a 32-half row
    const _Float16* gA = A + (size_t)m0 * GK;
    const _Float16* gB = Bt + (size_t)n0 * GK;

    // prologue: tile kt=0 into buffer 0
    {
        u32x4 a0 = *(const u32x4*)(gA + (size_t)lrow * GK + lseg);
        u32x4 a1 = *(const u32x4*)(gA + (size_t)(lrow + 64) * GK + lseg);
        u32x4 b0 = *(const u32x4*)(gB + (size_t)lrow * GK + lseg);
        u32x4 b1 = *(const u32x4*)(gB + (size_t)(lrow + 64) * GK + lseg);
        *(u32x4*)&sA[0][lrow * LDSS + lseg] = a0;
        *(u32x4*)&sA[0][(lrow + 64) * LDSS + lseg] = a1;
        *(u32x4*)&sB[0][lrow * LDSS + lseg] = b0;
        *(u32x4*)&sB[0][(lrow + 64) * LDSS + lseg] = b1;
    }
    __syncthreads();

    const int wid = tid >> 5, lane = tid & 31;
    const int l16 = lane & 15, g = lane >> 4;
    const int wm = (wid >> 1) * 32;   // wave M offset in block tile
    const int wn = (wid & 1) * 64;    // wave N offset in block tile

    v8f c[2][4];
#pragma unroll
    for (int mi = 0; mi < 2; ++mi)
#pragma unroll
        for (int ni = 0; ni < 4; ++ni) c[mi][ni] = (v8f)(0.f);

    const int KT = GK / 32;
    for (int kt = 0; kt < KT; ++kt) {
        const int cur = kt & 1;
        const bool more = (kt + 1) < KT;
        u32x4 pa0 = {}, pa1 = {}, pb0 = {}, pb1 = {};
        if (more) {
            const int ko = (kt + 1) * 32;
            pa0 = *(const u32x4*)(gA + (size_t)lrow * GK + ko + lseg);
            pa1 = *(const u32x4*)(gA + (size_t)(lrow + 64) * GK + ko + lseg);
            pb0 = *(const u32x4*)(gB + (size_t)lrow * GK + ko + lseg);
            pb1 = *(const u32x4*)(gB + (size_t)(lrow + 64) * GK + ko + lseg);
        }
        // gather WMMA fragments from LDS (CDNA5 documented layouts)
        const _Float16* sa = &sA[cur][0];
        const _Float16* sb = &sB[cur][0];
        union Frag { u32x4 q[2]; v16h h; } af[2], bf[4];
#pragma unroll
        for (int mi = 0; mi < 2; ++mi) {
            int r = wm + mi * 16 + l16;
            // A: VGPR j<4 holds K={2j,2j+1}+8g ; j>=4 holds K={2j+8,...}+8g
            af[mi].q[0] = *(const u32x4*)(sa + r * LDSS + 8 * g);
            af[mi].q[1] = *(const u32x4*)(sa + r * LDSS + 16 + 8 * g);
        }
#pragma unroll
        for (int ni = 0; ni < 4; ++ni) {
            int r = wn + ni * 16 + l16;
            // B: VGPR j<4 holds K={2j,2j+1}+8g ; j>=4 holds K={2j+8,...}+8g for column n=l16 (Bt is N-major)
            bf[ni].q[0] = *(const u32x4*)(sb + r * LDSS + 8 * g);
            bf[ni].q[1] = *(const u32x4*)(sb + r * LDSS + 16 + 8 * g);
        }
#pragma unroll
        for (int mi = 0; mi < 2; ++mi)
#pragma unroll
            for (int ni = 0; ni < 4; ++ni)
                c[mi][ni] = __builtin_amdgcn_wmma_f32_16x16x32_f16(
                    false, af[mi].h, false, bf[ni].h, (short)0, c[mi][ni], false, false);
        if (more) {
            const int nxt = cur ^ 1;
            *(u32x4*)&sA[nxt][lrow * LDSS + lseg] = pa0;
            *(u32x4*)&sA[nxt][(lrow + 64) * LDSS + lseg] = pa1;
            *(u32x4*)&sB[nxt][lrow * LDSS + lseg] = pb0;
            *(u32x4*)&sB[nxt][(lrow + 64) * LDSS + lseg] = pb1;
        }
        __syncthreads();
    }

    // epilogue: C/D layout -> element (r + 8g, l16) of each 16x16 subtile
#pragma unroll
    for (int mi = 0; mi < 2; ++mi)
#pragma unroll
        for (int ni = 0; ni < 4; ++ni)
#pragma unroll
            for (int r = 0; r < 8; ++r) {
                int m = m0 + wm + mi * 16 + r + 8 * g;
                int n = n0 + wn + ni * 16 + l16;
                float vv = c[mi][ni][r];
                if (RELU_F16_OUT) {
                    ((_Float16*)Cout)[(size_t)m * GN + n] = (_Float16)fmaxf(vv, 0.f);
                } else {
                    ((float*)Cout)[(size_t)m * GN + n] = vv;
                }
            }
}

// 9) out = LN(x1 + ff)
__global__ void k_ffn_ln(const float* __restrict__ x1f, const float* __restrict__ ff,
                         const float* __restrict__ g2, const float* __restrict__ b2,
                         float* __restrict__ out) {
    __shared__ float red[256];
    int bt = blockIdx.x;
    const float* xr = x1f + (size_t)bt * DD;
    const float* fr = ff + (size_t)bt * DD;
    float v[4];
#pragma unroll
    for (int j = 0; j < 4; ++j) {
        int c = threadIdx.x + j * 256;
        v[j] = xr[c] + fr[c];
    }
    float mean = block_sum256(v[0] + v[1] + v[2] + v[3], red) * (1.f / DD);
    float q = 0.f;
#pragma unroll
    for (int j = 0; j < 4; ++j) { float dd = v[j] - mean; q += dd * dd; }
    float var = block_sum256(q, red) * (1.f / DD);
    float sc = rsqrtf(var + LN_EPS);
#pragma unroll
    for (int j = 0; j < 4; ++j) {
        int c = threadIdx.x + j * 256;
        out[(size_t)bt * DD + c] = (v[j] - mean) * sc * g2[c] + b2[c];
    }
}

// ---------------------------------------------------------------------------
extern "C" void kernel_launch(void* const* d_in, const int* in_sizes, int n_in,
                              void* d_out, int out_size, void* d_ws, size_t ws_size,
                              hipStream_t stream) {
    (void)in_sizes; (void)n_in; (void)out_size; (void)ws_size;
    const float* x  = (const float*)d_in[0];
    const float* Wq = (const float*)d_in[1];
    const float* Wk = (const float*)d_in[2];
    const float* Wv = (const float*)d_in[3];
    const float* W1 = (const float*)d_in[4];
    const float* W2 = (const float*)d_in[5];
    const float* g1 = (const float*)d_in[6];
    const float* b1 = (const float*)d_in[7];
    const float* g2 = (const float*)d_in[8];
    const float* b2 = (const float*)d_in[9];
    float* out = (float*)d_out;
    char* ws = (char*)d_ws;

    // workspace layout (256B aligned offsets)
    constexpr size_t o_xsum  = 0;                                  // B*D f32
    constexpr size_t o_ksum  = o_xsum  + (size_t)BB * DD * 4;      // B*H*DH f32
    constexpr size_t o_vsum  = o_ksum  + (size_t)BB * HH * DHH * 4;
    constexpr size_t o_wqef  = o_vsum  + (size_t)BB * HH * DHH * 4; // B*H*D f32
    constexpr size_t o_rows  = o_wqef  + (size_t)BB * HH * DD * 4;  // B*H*T f32
    constexpr size_t o_x1f   = o_rows  + (size_t)BB * HH * TT * 4;  // M*D f32
    constexpr size_t o_x1h   = o_x1f   + (size_t)MM * DD * 4;       // M*D f16
    constexpr size_t o_h1h   = o_x1h   + (size_t)MM * DD * 2;       // M*D f16
    constexpr size_t o_ff    = o_h1h   + (size_t)MM * DD * 2;       // M*D f32
    constexpr size_t o_w1t   = o_ff    + (size_t)MM * DD * 4;       // D*D f16
    constexpr size_t o_w2t   = o_w1t   + (size_t)DD * DD * 2;       // D*D f16

    float*    xsum  = (float*)(ws + o_xsum);
    float*    ksum  = (float*)(ws + o_ksum);
    float*    vsum  = (float*)(ws + o_vsum);
    float*    wqeff = (float*)(ws + o_wqef);
    float*    rows  = (float*)(ws + o_rows);
    float*    x1f   = (float*)(ws + o_x1f);
    _Float16* x1h   = (_Float16*)(ws + o_x1h);
    _Float16* h1h   = (_Float16*)(ws + o_h1h);
    float*    ff    = (float*)(ws + o_ff);
    _Float16* w1t   = (_Float16*)(ws + o_w1t);
    _Float16* w2t   = (_Float16*)(ws + o_w2t);

    // attention path (algebraically reduced, exact f32)
    k_xsum   <<<(BB * DD) / 256, 256, 0, stream>>>(x, xsum);
    k_kvsum  <<<(BB * HH * DHH) / 256, 256, 0, stream>>>(xsum, Wk, Wv, ksum, vsum);
    k_wqeff  <<<(BB * HH * DD) / 256, 256, 0, stream>>>(Wq, ksum, wqeff);
    k_rowsum <<<(BB * HH * TT * 32) / 256, 256, 0, stream>>>(x, wqeff, rows);
    k_softmax<<<BB * HH, 256, 0, stream>>>(rows);
    k_attn_ln<<<BB * TT, 256, 0, stream>>>(x, rows, vsum, g1, b1, x1f, x1h);

    // weight convert + transpose for WMMA B operands
    dim3 tb(32, 8), tg(DD / 32, DD / 32);
    k_trans_h<<<tg, tb, 0, stream>>>(W1, w1t);
    k_trans_h<<<tg, tb, 0, stream>>>(W2, w2t);

    // FFN: h1 = relu(x1 @ W1) (f16 out), ff = h1 @ W2 (f32 out)
    dim3 gg(DD / 128, MM / 128);
    k_gemm<1><<<gg, 256, 0, stream>>>(x1h, w1t, (void*)h1h);
    k_gemm<0><<<gg, 256, 0, stream>>>(h1h, w2t, (void*)ff);

    // out = LN(x1 + ff)
    k_ffn_ln<<<BB * TT, 256, 0, stream>>>(x1f, ff, g2, b2, out);
}
